// EncoderLayer_2035814498815
// MI455X (gfx1250) — matrix operand order repaired and checked by reading, not compiled, not run on hardware
//
#include <hip/hip_runtime.h>
#include <hip/hip_bf16.h>
#include <math.h>

typedef __attribute__((ext_vector_type(16))) __bf16 v16bf;
typedef __attribute__((ext_vector_type(8)))  __bf16 bf16x8;
typedef __attribute__((ext_vector_type(8)))  float  v8f;
typedef __attribute__((ext_vector_type(8)))  float  f32x8;
typedef __attribute__((ext_vector_type(4)))  int    v4i_t;

#if __has_builtin(__builtin_amdgcn_global_load_async_to_lds_b128)
#define ASYNC_STAGE 1
#endif

namespace {
constexpr int kB = 4, kS = 1024, kD = 512, kH = 8, kF = 2048;
constexpr int TM = 128, TN = 128, TK = 32;
constexpr int LP = 40;            // LDS row pitch (bf16): 80B, multiple of 16B
constexpr int QCHUNK = 128;
}

// ---------------------------------------------------------------- utilities
__global__ void cast_bf16_k(const float* __restrict__ s, __bf16* __restrict__ d, int n) {
  int i = (blockIdx.x * 256 + threadIdx.x) * 8;
  if (i < n) {
    f32x8 v = *(const f32x8*)(s + i);
    bf16x8 o;
#pragma unroll
    for (int e = 0; e < 8; ++e) o[e] = (__bf16)v[e];
    *(bf16x8*)(d + i) = o;
  }
}

__global__ void fill0_k(float* __restrict__ p, int n) {
  int i = blockIdx.x * 256 + threadIdx.x;
  if (i < n) p[i] = 0.0f;
}

__device__ __forceinline__ float mask_val(int mt, int q, int k) {
  if (mt == 0) return 0.0f;
  int dq = (q >> 5) - (k >> 5); dq = dq < 0 ? -dq : dq;
  int dr = (q & 31) - (k & 31); dr = dr < 0 ? -dr : dr;
  bool inside = (dq <= 3) && (dr <= 3);
  bool blocked = (mt == 1) ? !inside : inside;   // 1 = m_rev, 2 = m_fwd
  return blocked ? -3.0e38f : 0.0f;
}

__device__ __forceinline__ void async_copy16(const __bf16* g, __bf16* l) {
#if defined(ASYNC_STAGE)
  __builtin_amdgcn_global_load_async_to_lds_b128(
      (__attribute__((address_space(1))) v4i_t*)g,
      (__attribute__((address_space(3))) v4i_t*)l, 0, 0);
#else
  *(bf16x8*)l = *(const bf16x8*)g;
#endif
}

__device__ __forceinline__ void wait_async_lds() {
#if defined(ASYNC_STAGE)
#if __has_builtin(__builtin_amdgcn_s_wait_asynccnt)
  __builtin_amdgcn_s_wait_asynccnt(0);
#else
  asm volatile("s_wait_asynccnt 0" ::: "memory");
#endif
#endif
}

// ---------------------------------------------------------------- GEMM (bf16 WMMA)
// D = alpha*(A x B^T) + bias, optional GELU, optional gated accumulate into C,
// optional transposed bf16 store (flags&8: Cbf[col][row], vectorized).
// A: M x K row major (lda). Bm: N x K row major (ldb) — always "B^T supplied".
// Batched: z -> (zb = z/Hdiv, zh = z%Hdiv), per-operand (batch, head) strides.
// Block tile 128x128, 8 waves as 2(M) x 4(N), wave tile 64x32 -> 8 WMMA/K-step.
__global__ __launch_bounds__(256) void gemm_bf16_k(
    const __bf16* __restrict__ A, long long sAb, long long sAh, int lda,
    const __bf16* __restrict__ Bm, long long sBb, long long sBh, int ldb,
    const float* __restrict__ bias,
    float* __restrict__ C, long long sCb, long long sCh, int ldc,
    __bf16* __restrict__ Cbf, int ldct, long long sCtb,
    const float* __restrict__ gate, long long sGb, int gidx,
    int M, int N, int K, int Hdiv, float alpha, int flags)
{
  const int z  = blockIdx.z;
  const int zb = z / Hdiv, zh = z % Hdiv;
  A  += (size_t)(zb * sAb + zh * sAh);
  Bm += (size_t)(zb * sBb + zh * sBh);
  const long long coff = zb * sCb + zh * sCh;

  // double-buffered tiles, both k-contiguous per row so fragments are 2x b128
  __shared__ __align__(16) __bf16 At[2][TM][LP];   // [m][k]
  __shared__ __align__(16) __bf16 Bt[2][TN][LP];   // [n][k]

  const int tid   = threadIdx.x;
  const int mBase = blockIdx.y * TM;
  const int nBase = blockIdx.x * TN;
  const int wave  = tid >> 5;
  const int lane  = tid & 31;
  const int hl    = lane >> 4;           // lane half (0/1)
  const int l16   = lane & 15;
  const int wm    = (wave >> 2) * 64;    // wave tile origin in M (0/64)
  const int wn    = (wave & 3) * 32;     // wave tile origin in N (0..96)

  v8f acc[4][2] = {};

  // stage one 128x32 tile pair; 512 16B chunks each, 2 per thread per tile
  auto stage = [&](int k0, int buf) {
#pragma unroll
    for (int t = 0; t < 2; ++t) {
      int i = tid + t * 256;
      int r = i >> 2, kc = (i & 3) * 8;
      // A rows are always in bounds (all M are multiples of 128)
      async_copy16(A + (size_t)(mBase + r) * lda + (k0 + kc), &At[buf][r][kc]);
      {
        int gn = nBase + r;
        __bf16* l = &Bt[buf][r][kc];
        if (gn < N) {
          async_copy16(Bm + (size_t)gn * ldb + (k0 + kc), l);
        } else {
          bf16x8 zv = {};
          *(bf16x8*)l = zv;     // zero-fill (only P@V's N=64 hits this)
        }
      }
    }
  };

  const int nk = K / TK;
  stage(0, 0);
  wait_async_lds();
  __syncthreads();

  for (int ki = 0; ki < nk; ++ki) {
    const int cur = ki & 1;
    if (ki + 1 < nk) stage((ki + 1) * TK, cur ^ 1);   // overlaps WMMA below

    // fragments: ISA 16-bit A(16x32)/B(32x16) lane maps, 2x ds_load_b128 each
    v16bf af[4], bfr[2];
#pragma unroll
    for (int i = 0; i < 4; ++i) {
      int mA = wm + i * 16 + l16;
      bf16x8 lo = *(const bf16x8*)&At[cur][mA][8 * hl];
      bf16x8 hi = *(const bf16x8*)&At[cur][mA][16 + 8 * hl];
      af[i] = __builtin_shufflevector(lo, hi, 0, 1, 2, 3, 4, 5, 6, 7,
                                      8, 9, 10, 11, 12, 13, 14, 15);
    }
#pragma unroll
    for (int j = 0; j < 2; ++j) {
      int nB = wn + j * 16 + l16;
      bf16x8 lo = *(const bf16x8*)&Bt[cur][nB][8 * hl];
      bf16x8 hi = *(const bf16x8*)&Bt[cur][nB][16 + 8 * hl];
      bfr[j] = __builtin_shufflevector(lo, hi, 0, 1, 2, 3, 4, 5, 6, 7,
                                       8, 9, 10, 11, 12, 13, 14, 15);
    }

#pragma unroll
    for (int i = 0; i < 4; ++i)
#pragma unroll
      for (int j = 0; j < 2; ++j)
        acc[i][j] = __builtin_amdgcn_wmma_f32_16x16x32_bf16(
            false, af[i], false, bfr[j], (short)0, acc[i][j], false, false);

    wait_async_lds();       // next tile landed (own wave's asyncs)
    __syncthreads();        // ... and everyone else's
  }

  // epilogue: C layout — lane l16 = column, VGPR r -> row 8*hl + r
#pragma unroll
  for (int i = 0; i < 4; ++i) {
#pragma unroll
    for (int j = 0; j < 2; ++j) {
      int gcol = nBase + wn + j * 16 + l16;
      if (gcol >= N) continue;
      float bv = bias ? bias[gcol] : 0.0f;
      if (flags & 8) {
        // transposed bf16 store: rows are contiguous -> one 16B store
        int growBase = mBase + wm + i * 16 + 8 * hl;
        bf16x8 o;
#pragma unroll
        for (int r = 0; r < 8; ++r) o[r] = (__bf16)(acc[i][j][r] * alpha + bv);
        *(bf16x8*)(Cbf + (size_t)zb * sCtb + (size_t)gcol * ldct + growBase) = o;
        continue;
      }
#pragma unroll
      for (int r = 0; r < 8; ++r) {
        int grow = mBase + wm + i * 16 + 8 * hl + r;
        if (grow >= M) continue;
        float v = acc[i][j][r] * alpha + bv;
        if (flags & 2) v = 0.5f * v * (1.0f + erff(v * 0.70710678118654752f));
        size_t off = (size_t)(coff + (long long)grow * ldc + gcol);
        if (flags & 4) {
          float gv = gate[(size_t)zb * sGb + (size_t)grow * 5 + gidx];
          C[off] += gv * v;
        } else {
          if (C)   C[off]   = v;
          if (Cbf) Cbf[off] = (__bf16)v;
        }
      }
    }
  }
}

// ---------------------------------------------------------------- softmax (+mask)
// rows are [bh][128][S]; one 256-thread block per row.
__global__ __launch_bounds__(256) void softmax_k(
    const float* __restrict__ sc, __bf16* __restrict__ pr,
    int S, int qbase, int masktype)
{
  const int row = blockIdx.x;
  const int q   = qbase + (row & (QCHUNK - 1));
  const float* srow = sc + (size_t)row * S;
  __bf16*      prow = pr + (size_t)row * S;
  __shared__ float red[256];
  const int tid = threadIdx.x;

  float mx = -3.0e38f;
  for (int k = tid; k < S; k += 256)
    mx = fmaxf(mx, srow[k] + mask_val(masktype, q, k));
  red[tid] = mx; __syncthreads();
  for (int st = 128; st > 0; st >>= 1) {
    if (tid < st) red[tid] = fmaxf(red[tid], red[tid + st]);
    __syncthreads();
  }
  mx = red[0]; __syncthreads();

  float sum = 0.0f;
  for (int k = tid; k < S; k += 256)
    sum += __expf(srow[k] + mask_val(masktype, q, k) - mx);
  red[tid] = sum; __syncthreads();
  for (int st = 128; st > 0; st >>= 1) {
    if (tid < st) red[tid] += red[tid + st];
    __syncthreads();
  }
  float inv = 1.0f / red[0];

  for (int k = tid; k < S; k += 256) {
    float e = __expf(srow[k] + mask_val(masktype, q, k) - mx);
    prow[k] = (__bf16)(e * inv);
  }
}

// ---------------------------------------------------------------- layernorm (+residual)
__global__ __launch_bounds__(256) void layernorm_k(
    const float* __restrict__ xin, const float* __restrict__ add,
    const float* __restrict__ g, const float* __restrict__ b,
    float* __restrict__ out, __bf16* __restrict__ outbf, int width)
{
  const int row = blockIdx.x;
  const float* xr = xin + (size_t)row * width;
  const float* ar = add ? add + (size_t)row * width : nullptr;
  __shared__ float red[256];
  const int tid = threadIdx.x;

  float s = 0.0f;
  for (int i = tid; i < width; i += 256) s += xr[i] + (ar ? ar[i] : 0.0f);
  red[tid] = s; __syncthreads();
  for (int st = 128; st > 0; st >>= 1) {
    if (tid < st) red[tid] += red[tid + st];
    __syncthreads();
  }
  float mean = red[0] / (float)width; __syncthreads();

  float s2 = 0.0f;
  for (int i = tid; i < width; i += 256) {
    float v = xr[i] + (ar ? ar[i] : 0.0f) - mean;
    s2 += v * v;
  }
  red[tid] = s2; __syncthreads();
  for (int st = 128; st > 0; st >>= 1) {
    if (tid < st) red[tid] += red[tid + st];
    __syncthreads();
  }
  float inv = rsqrtf(red[0] / (float)width + 1e-5f);

  for (int i = tid; i < width; i += 256) {
    float v = (xr[i] + (ar ? ar[i] : 0.0f) - mean) * inv * g[i] + b[i];
    out[(size_t)row * width + i] = v;
    if (outbf) outbf[(size_t)row * width + i] = (__bf16)v;
  }
}

// ---------------------------------------------------------------- host side
static inline void launch_gemm(hipStream_t st,
    const __bf16* A, long long sAb, long long sAh, int lda,
    const __bf16* Bm, long long sBb, long long sBh, int ldb,
    const float* bias,
    float* C, long long sCb, long long sCh, int ldc,
    __bf16* Cbf, int ldct, long long sCtb,
    const float* gate, long long sGb, int gidx,
    int M, int N, int K, int batches, int Hdiv, float alpha, int flags)
{
  dim3 grid((N + TN - 1) / TN, (M + TM - 1) / TM, batches);
  gemm_bf16_k<<<grid, 256, 0, st>>>(A, sAb, sAh, lda, Bm, sBb, sBh, ldb, bias,
                                    C, sCb, sCh, ldc, Cbf, ldct, sCtb,
                                    gate, sGb, gidx, M, N, K, Hdiv, alpha, flags);
}

extern "C" void kernel_launch(void* const* d_in, const int* in_sizes, int n_in,
                              void* d_out, int out_size, void* d_ws, size_t ws_size,
                              hipStream_t stream) {
  (void)in_sizes; (void)n_in; (void)out_size; (void)ws_size;

  const float* x    = (const float*)d_in[0];
  const float* ref  = (const float*)d_in[1];
  const float* gate = (const float*)d_in[2];
  const float* aw[5], *ab[5], *aow[5], *aob[5];
  for (int a = 0; a < 5; ++a) {
    aw[a]  = (const float*)d_in[3 + 4 * a + 0];
    ab[a]  = (const float*)d_in[3 + 4 * a + 1];
    aow[a] = (const float*)d_in[3 + 4 * a + 2];
    aob[a] = (const float*)d_in[3 + 4 * a + 3];
  }
  const float* ln1g = (const float*)d_in[23];
  const float* ln1b = (const float*)d_in[24];
  const float* ln2g = (const float*)d_in[25];
  const float* ln2b = (const float*)d_in[26];
  const float* fc1w = (const float*)d_in[27];
  const float* fc1b = (const float*)d_in[28];
  const float* fc2w = (const float*)d_in[29];
  const float* fc2b = (const float*)d_in[30];
  float* out = (float*)d_out;

  // ---- workspace carve-out
  size_t off = 0;
  auto alloc = [&](size_t bytes) -> void* {
    off = (off + 255) & ~(size_t)255;
    void* p = (char*)d_ws + off;
    off += bytes;
    return p;
  };
  const size_t NX = (size_t)kB * kS * kD;           // 2,097,152
  __bf16* xbf    = (__bf16*)alloc(2 * NX);
  __bf16* refbf  = (__bf16*)alloc(2 * 2 * NX);
  __bf16* wallbf = (__bf16*)alloc(2ull * 5 * 3 * kD * kD);
  __bf16* owallbf= (__bf16*)alloc(2ull * 5 * kD * kD);
  __bf16* fc1bf  = (__bf16*)alloc(2ull * kF * kD);
  __bf16* fc2bf  = (__bf16*)alloc(2ull * kF * kD);
  __bf16* qbf    = (__bf16*)alloc(2 * NX);
  __bf16* kbf    = (__bf16*)alloc(2 * NX);
  __bf16* vbfT   = (__bf16*)alloc(2 * NX);          // V^T: [B][D][S]
  float*  scores = (float*) alloc(4ull * kB * kH * QCHUNK * kS);   // 16 MB slab (L2-resident)
  __bf16* probs  = (__bf16*)alloc(2ull * kB * kH * QCHUNK * kS);   // 8 MB slab
  __bf16* attnbf = (__bf16*)alloc(2 * NX);
  float*  newx   = (float*) alloc(4 * NX);
  float*  hbuf   = (float*) alloc(4 * NX);
  __bf16* hbf    = (__bf16*)alloc(2 * NX);
  __bf16* ffn1bf = (__bf16*)alloc(2ull * kB * kS * kF);
  float*  ybuf   = (float*) alloc(4 * NX);

  auto cast = [&](const float* s, __bf16* d, size_t n) {
    cast_bf16_k<<<(unsigned)((n / 8 + 255) / 256), 256, 0, stream>>>(s, d, (int)n);
  };

  // ---- one-time bf16 conversions
  cast(x,   xbf,   NX);
  cast(ref, refbf, 2 * NX);
  for (int a = 0; a < 5; ++a) {
    cast(aw[a],  wallbf  + (size_t)a * 3 * kD * kD, (size_t)3 * kD * kD);
    cast(aow[a], owallbf + (size_t)a * kD * kD,     (size_t)kD * kD);
  }
  cast(fc1w, fc1bf, (size_t)kF * kD);
  cast(fc2w, fc2bf, (size_t)kF * kD);

  fill0_k<<<(unsigned)((NX + 255) / 256), 256, 0, stream>>>(newx, (int)NX);

  // ---- the six attention passes (mca runs twice; gating is linear)
  struct Pass { int w; const __bf16* ksrc; long long sk; const __bf16* vsrc; long long sv; int mask; int g; };
  const long long SD  = (long long)kS * kD;
  const long long SD2 = 2 * SD;
  Pass passes[6] = {
    {0, refbf,               SD2, refbf,               SD2, 1, 0},   // mca r=0, m_rev
    {0, refbf + (size_t)SD,  SD2, refbf + (size_t)SD,  SD2, 1, 0},   // mca r=1, m_rev
    {1, refbf + (size_t)SD,  SD2, refbf + (size_t)SD,  SD2, 0, 1},   // ca, ref_last, no mask
    {2, xbf,                 SD,  xbf,                 SD,  2, 2},   // msa, m_fwd
    {3, refbf + (size_t)SD,  SD2, xbf,                 SD,  2, 3},   // nsa: k=ref_last, v=x, m_fwd
    {4, xbf,                 SD,  xbf,                 SD,  0, 4},   // sa, no mask
  };

  const float inv_sqrt_hd = 0.125f;  // 1/sqrt(64)
  for (int p = 0; p < 6; ++p) {
    const Pass& ps = passes[p];
    const __bf16* wset  = wallbf  + (size_t)ps.w * 3 * kD * kD;
    const __bf16* owset = owallbf + (size_t)ps.w * kD * kD;
    const float*  bset  = ab[ps.w];
    const float*  obset = aob[ps.w];

    // Q/K/V projections: y = t @ W^T + b, bf16 out (V written transposed)
    launch_gemm(stream, xbf, SD, 0, kD, wset, 0, 0, kD, bset,
                nullptr, 0, 0, kD, qbf, 0, 0, nullptr, 0, 0,
                kS, kD, kD, kB, 1, 1.0f, 0);
    launch_gemm(stream, ps.ksrc, ps.sk, 0, kD, wset + (size_t)kD * kD, 0, 0, kD, bset + kD,
                nullptr, 0, 0, kD, kbf, 0, 0, nullptr, 0, 0,
                kS, kD, kD, kB, 1, 1.0f, 0);
    launch_gemm(stream, ps.vsrc, ps.sv, 0, kD, wset + (size_t)2 * kD * kD, 0, 0, kD, bset + 2 * kD,
                nullptr, 0, 0, kD, vbfT, kS, (long long)kD * kS, nullptr, 0, 0,
                kS, kD, kD, kB, 1, 1.0f, 8 /*transposed bf16 store*/);

    // attention core, q-chunked so the score slab stays in L2
    for (int qc = 0; qc < kS / QCHUNK; ++qc) {
      int qoff = qc * QCHUNK;
      // scores[bh, 128, S] = (Q/sqrt(hd)) @ K^T
      launch_gemm(stream,
                  qbf + (size_t)qoff * kD, SD, 64, kD,
                  kbf,                     SD, 64, kD,
                  nullptr,
                  scores, (long long)kH * QCHUNK * kS, (long long)QCHUNK * kS, kS,
                  nullptr, 0, 0, nullptr, 0, 0,
                  QCHUNK, kS, 64, kB * kH, kH, inv_sqrt_hd, 0);
      softmax_k<<<kB * kH * QCHUNK, 256, 0, stream>>>(scores, probs, kS, qoff, ps.mask);
      // attn[b, q, h*64:d] = P @ V  (B operand = V^T, so it's the NxK path too)
      launch_gemm(stream,
                  probs, (long long)kH * QCHUNK * kS, (long long)QCHUNK * kS, kS,
                  vbfT,  (long long)kD * kS, (long long)64 * kS, kS,
                  nullptr,
                  nullptr, SD, 64, kD,
                  attnbf + (size_t)qoff * kD, 0, 0, nullptr, 0, 0,
                  QCHUNK, 64, kS, kB * kH, kH, 1.0f, 0);
    }

    // output projection, fused gated accumulate into new_x
    launch_gemm(stream, attnbf, SD, 0, kD, owset, 0, 0, kD, obset,
                newx, SD, 0, kD, nullptr, 0, 0,
                gate, (long long)kS * 5, ps.g,
                kS, kD, kD, kB, 1, 1.0f, 4);
  }

  // ---- FFN + layernorms
  layernorm_k<<<kB * kS, 256, 0, stream>>>(newx, nullptr, ln1g, ln1b, hbuf, hbf, kD);
  launch_gemm(stream, hbf, 0, 0, kD, fc1bf, 0, 0, kD, fc1b,
              nullptr, 0, 0, kF, ffn1bf, 0, 0, nullptr, 0, 0,
              kB * kS, kF, kD, 1, 1, 1.0f, 2 /*GELU*/);
  launch_gemm(stream, ffn1bf, 0, 0, kF, fc2bf, 0, 0, kF, fc2b,
              ybuf, 0, 0, kD, nullptr, 0, 0, nullptr, 0, 0,
              kB * kS, kD, kF, 1, 1, 1.0f, 0);
  layernorm_k<<<kB * kS, 256, 0, stream>>>(hbuf, ybuf, ln2g, ln2b, out, nullptr, kD);
}
